// STAtt_31026843747109
// MI455X (gfx1250) — matrix operand order repaired and checked by reading, not compiled, not run on hardware
//
#include <hip/hip_runtime.h>
#include <hip/hip_bf16.h>
#include <math.h>
#include <stdint.h>

// ---------------------------------------------------------------------------
// MI455X (gfx1250) implementation.
// Compute-dominant part: three 3x3 convs + one 1x1 conv -> implicit GEMM on
// v_wmma_f32_16x16x32_bf16 (wave32, 16x16 tiles). f32 global data is converted
// to bf16 while staging into LDS; f32 accumulation throughout.
// The 1x1 post-conv additionally demonstrates the Tensor Data Mover path:
// tensor_load_to_lds + s_wait_tensorcnt for global->LDS tile DMA.
// ---------------------------------------------------------------------------

typedef __bf16 bf16;
typedef __attribute__((ext_vector_type(16))) __bf16 v16bf;
typedef __attribute__((ext_vector_type(8)))  __bf16 bf16x8;
typedef __attribute__((ext_vector_type(8)))  float  v8f;
typedef __attribute__((ext_vector_type(4)))  unsigned int v4u;
typedef __attribute__((ext_vector_type(8)))  int v8i;
typedef __attribute__((ext_vector_type(4)))  int v4i;

#define NS   128          // S*B samples for the 3x3 convs
#define CHW  131072       // 128 * 32 * 32
#define HW   1024

__device__ __forceinline__ float sigm(float x) { return 1.f / (1.f + expf(-x)); }

// ---------------------------------------------------------------------------
// TDM: issue a 2D-tile tensor_load_to_lds. All args uniform (SGPR descriptor).
// D# Group0: [1:0]=count=1, [63:32]=lds byte addr, [120:64]=global byte addr,
//            [127:126]=type=2.  Group1: data_size=4B, tensor/tile dims+stride.
// Dims/strides in data_size units; tile==tensor extent so nothing is OOB.
// This toolchain (clang-23 / therock-10.0) uses the 6-arg builtin:
//   (uint32x4 g0, int32x8 g1, int32x4 g2, int32x4 g3, int32x8 extra, i32 cpol)
// Groups 2/3 + extra are zero (tile_dim2/3/4 = 0 -> 2D transfer only).
// ---------------------------------------------------------------------------
__device__ __forceinline__ void tdm_load_2d(unsigned lds_off, const void* gptr,
                                            unsigned td0, unsigned td1,
                                            unsigned tile0, unsigned tile1,
                                            unsigned stride0)
{
#if __has_builtin(__builtin_amdgcn_tensor_load_to_lds)
    uint64_t ga = (uint64_t)(uintptr_t)gptr;
    v4u g0;
    g0.x = 1u;                                   // count=1, user mode
    g0.y = lds_off;                              // LDS dest byte address
    g0.z = (unsigned)ga;                         // global addr [31:0]
    g0.w = (unsigned)(ga >> 32) | (2u << 30);    // global addr [56:32] | type=2
    v8i g1;
    g1[0] = (int)(2u << 16);                     // wg_mask=0, data_size=4B
    g1[1] = (int)((td0 & 0xFFFFu) << 16);        // tensor_dim0[15:0]
    g1[2] = (int)((td0 >> 16) | ((td1 & 0xFFFFu) << 16));
    g1[3] = (int)((td1 >> 16) | (tile0 << 16));  // tile_dim0
    g1[4] = (int)tile1;                          // tile_dim1 (tile_dim2 = 0)
    g1[5] = (int)stride0;                        // tensor_dim0_stride[31:0]
    g1[6] = 0;                                   // stride0 hi | stride1 lo
    g1[7] = 0;
    v4i z4 = {0, 0, 0, 0};                       // groups 2/3 unused (2D)
    v8i z8 = {0, 0, 0, 0, 0, 0, 0, 0};
    __builtin_amdgcn_tensor_load_to_lds(g0, g1, z4, z4, z8, 0);
#else
    (void)lds_off; (void)gptr; (void)td0; (void)td1;
    (void)tile0; (void)tile1; (void)stride0;
#endif
}

// ---------------------------------------------------------------------------
// Implicit-GEMM conv: out[n, co, y, x] = sum_{ci,dy,dx} w[co,ci,dy,dx] *
//                                        in[n, ci, y+dy-H, x+dx-H]  (+pos fused)
// Workgroup: 256 thr = 8 waves. Tile: M=128 (all Cout) x N=128 (4 image rows).
// Wave w owns rows [16w,16w+16) x N=128 -> 8 wmma accumulators.
// LDS: wtile[tap][co][ci32] (bf16), itile[row][col][ci32] (bf16, halo'd).
// ---------------------------------------------------------------------------
template <int CIN, int TAPS>
__global__ __launch_bounds__(256)
void conv_wmma_kernel(const float* __restrict__ in,    // [NSMP][CIN][32][32]
                      const float* __restrict__ wgt,   // [128][CIN][TAPS][TAPS]
                      float* __restrict__ out,         // [NSMP][128][32][32]
                      const float* __restrict__ posadd) // [8][CIN] or nullptr
{
    constexpr int T2   = TAPS * TAPS;
    constexpr int HALO = TAPS / 2;
    constexpr int ROWS = 4 + 2 * HALO;
    constexpr int COLS = 32 + 2 * HALO;

    extern __shared__ char smem[];
    bf16* wtile = reinterpret_cast<bf16*>(smem);        // T2*128*32
    bf16* itile = wtile + T2 * 128 * 32;                // ROWS*COLS*32

    const int tid  = threadIdx.x;
    const int lane = tid & 31;
    const int wv   = tid >> 5;        // 0..7
    const int half = lane >> 4;       // 0/1
    const int l16  = lane & 15;

    const int sample = blockIdx.y;
    const int y0     = blockIdx.x * 4;      // 4 image rows per N-tile

    const float* inS  = in  + (size_t)sample * CIN * HW;
    float*       outS = out + (size_t)sample * 128 * HW;
    const int    sidx = sample >> 4;        // temporal step (sample = s*16+b)

    v8f acc[8];
    #pragma unroll
    for (int i = 0; i < 8; ++i)
        #pragma unroll
        for (int j = 0; j < 8; ++j) acc[i][j] = 0.f;

    #pragma unroll 1
    for (int ci0 = 0; ci0 < CIN; ci0 += 32) {
        __syncthreads();
        // ---- stage input tile (f32 -> bf16), fused positional add for v ----
        for (int idx = tid; idx < 32 * ROWS * COLS; idx += 256) {
            int xc = idx % COLS;
            int t  = idx / COLS;
            int r  = t % ROWS;
            int ci = t / ROWS;
            int y  = y0 + r - HALO;
            int x  = xc - HALO;
            float v = 0.f;
            if (y >= 0 && y < 32 && x >= 0 && x < 32) {
                v = inS[(size_t)(ci0 + ci) * HW + y * 32 + x];
                if (posadd) v += posadd[sidx * CIN + ci0 + ci];
            }
            itile[(r * COLS + xc) * 32 + ci] = (bf16)v;
        }
        // ---- stage weights (f32 -> bf16): wtile[tap][co][ci] ----
        for (int idx = tid; idx < T2 * 128 * 32; idx += 256) {
            int tap = idx % T2;
            int t   = idx / T2;
            int ci  = t % 32;
            int co  = t / 32;
            wtile[(tap * 128 + co) * 32 + ci] =
                (bf16)wgt[((size_t)co * CIN + ci0 + ci) * T2 + tap];
        }
        __syncthreads();

        // prefetch next chunk (gfx1250 global_prefetch_b8 path)
        if (ci0 + 32 < CIN)
            __builtin_prefetch(inS + (size_t)(ci0 + 32 + lane) * HW + y0 * 32, 0, 1);

        // ---- compute: taps x 8 N-tiles of wmma ----
        #pragma unroll 1
        for (int tap = 0; tap < T2; ++tap) {
            const int dy = tap / TAPS, dx = tap % TAPS;
            // A fragment: lane<16 -> K 0..15; lane>=16 -> K 16..31 (contig)
            const int co = wv * 16 + l16;
            const bf16* wr = &wtile[(tap * 128 + co) * 32];
            bf16x8 a0 = *(const bf16x8*)(wr + half * 16);
            bf16x8 a1 = *(const bf16x8*)(wr + half * 16 + 8);
            v16bf a = __builtin_shufflevector(a0, a1,
                        0,1,2,3,4,5,6,7,8,9,10,11,12,13,14,15);
            #pragma unroll
            for (int nt = 0; nt < 8; ++nt) {
                // B fragment: lane<16 -> K 0..15, lane>=16 -> K 16..31 (contig)
                int n  = nt * 16 + l16;
                int ry = n >> 5, x = n & 31;
                const bf16* br =
                    &itile[((ry + dy) * COLS + (x + dx)) * 32 + half * 16];
                v16bf b = *(const v16bf*)br;
                acc[nt] = __builtin_amdgcn_wmma_f32_16x16x32_bf16(
                    false, a, false, b, (short)0, acc[nt], false, false);
            }
        }
    }

    // ---- epilogue: C/D layout (VGPR j: M=j lanes0-15 / M=j+8 lanes16-31) ----
    #pragma unroll
    for (int nt = 0; nt < 8; ++nt) {
        int n = nt * 16 + l16;
        int p = y0 * 32 + n;
        #pragma unroll
        for (int j = 0; j < 8; ++j) {
            int co = wv * 16 + j + half * 8;
            outS[(size_t)co * HW + p] = acc[nt][j];
        }
    }
}

// ---------------------------------------------------------------------------
// 1x1 conv (CIN=128) with TDM double-stage: tensor_load_to_lds pulls the f32
// input tile [ci32 x p128] and weight slab [co128 x ci32] into LDS, wave 0
// waits on TENSORcnt, then all threads convert to the bf16 WMMA layouts.
// LDS (dynamic, group-segment offset 0):
//   [    0, 16K) istage f32 [32][128]
//   [16384, 32K) wstage f32 [128][32]
//   [32768, 40K) wtile bf16 [128][32]
//   [40960, 48K) itile bf16 [128][32]
// ---------------------------------------------------------------------------
__global__ __launch_bounds__(256)
void conv1x1_wmma_tdm_kernel(const float* __restrict__ in,   // [16][128][32][32]
                             const float* __restrict__ wgt,  // [128][128]
                             float* __restrict__ out)        // [16][128][32][32]
{
    extern __shared__ char smem[];
    float* istage = (float*)(smem);
    float* wstage = (float*)(smem + 16384);
    bf16*  wtile  = (bf16*)(smem + 32768);
    bf16*  itile  = (bf16*)(smem + 40960);

    const int tid  = threadIdx.x;
    const int lane = tid & 31;
    const int wv   = tid >> 5;
    const int half = lane >> 4;
    const int l16  = lane & 15;

    const int sample = blockIdx.y;           // 0..15
    const int y0     = blockIdx.x * 4;       // 4 image rows per N-tile

    const float* inS  = in  + (size_t)sample * 128 * HW;
    float*       outS = out + (size_t)sample * 128 * HW;

    v8f acc[8];
    #pragma unroll
    for (int i = 0; i < 8; ++i)
        #pragma unroll
        for (int j = 0; j < 8; ++j) acc[i][j] = 0.f;

    #pragma unroll 1
    for (int ci0 = 0; ci0 < 128; ci0 += 32) {
        __syncthreads();
#if __has_builtin(__builtin_amdgcn_tensor_load_to_lds)
        if (wv == 0) {                       // wave-uniform branch; TDM is per-wave
            // input tile: 32 rows (ci), 128 contiguous f32 (4 image rows)
            tdm_load_2d(0u, inS + (size_t)ci0 * HW + y0 * 32,
                        /*td0*/128u, /*td1*/32u, /*tile0*/128u, /*tile1*/32u,
                        /*stride0*/1024u);
            // weight slab: 128 rows (co), 32 contiguous f32 (ci chunk)
            tdm_load_2d(16384u, wgt + ci0,
                        /*td0*/32u, /*td1*/128u, /*tile0*/32u, /*tile1*/128u,
                        /*stride0*/128u);
            __builtin_amdgcn_s_wait_tensorcnt(0);
        }
#else
        for (int idx = tid; idx < 4096; idx += 256) {
            int ci = idx >> 7, p = idx & 127;
            istage[idx] = inS[(size_t)(ci0 + ci) * HW + y0 * 32 + p];
            wstage[idx] = wgt[(size_t)(idx >> 5) * 128 + ci0 + (idx & 31)];
        }
#endif
        __syncthreads();
        // convert staged f32 -> bf16 WMMA layouts
        for (int idx = tid; idx < 4096; idx += 256) {
            int ci = idx >> 7, p = idx & 127;
            itile[p * 32 + ci] = (bf16)istage[idx];   // [p][ci]
            wtile[idx]         = (bf16)wstage[idx];   // [co][ci] (same layout)
        }
        __syncthreads();

        // A fragment
        const int co = wv * 16 + l16;
        const bf16* wr = &wtile[co * 32];
        bf16x8 a0 = *(const bf16x8*)(wr + half * 16);
        bf16x8 a1 = *(const bf16x8*)(wr + half * 16 + 8);
        v16bf a = __builtin_shufflevector(a0, a1,
                    0,1,2,3,4,5,6,7,8,9,10,11,12,13,14,15);
        #pragma unroll
        for (int nt = 0; nt < 8; ++nt) {
            int n = nt * 16 + l16;
            const bf16* br = &itile[n * 32 + half * 16];
            v16bf b = *(const v16bf*)br;
            acc[nt] = __builtin_amdgcn_wmma_f32_16x16x32_bf16(
                false, a, false, b, (short)0, acc[nt], false, false);
        }
    }

    #pragma unroll
    for (int nt = 0; nt < 8; ++nt) {
        int n = nt * 16 + l16;
        int p = y0 * 32 + n;
        #pragma unroll
        for (int j = 0; j < 8; ++j) {
            int co = wv * 16 + j + half * 8;
            outS[(size_t)co * HW + p] = acc[nt][j];
        }
    }
}

// --------------------- GroupNorm(1) per sample, optional SiLU ---------------
__global__ __launch_bounds__(256)
void gn_kernel(const float* __restrict__ x, const float* __restrict__ g,
               const float* __restrict__ bt, float* __restrict__ y, int silu)
{
    __shared__ float rs[256], rq[256];
    const int n = blockIdx.x;
    const float* xp = x + (size_t)n * CHW;
    float s = 0.f, q = 0.f;
    for (int i = threadIdx.x; i < CHW; i += 256) {
        float v = xp[i]; s += v; q += v * v;
    }
    rs[threadIdx.x] = s; rq[threadIdx.x] = q; __syncthreads();
    for (int st = 128; st > 0; st >>= 1) {
        if (threadIdx.x < st) {
            rs[threadIdx.x] += rs[threadIdx.x + st];
            rq[threadIdx.x] += rq[threadIdx.x + st];
        }
        __syncthreads();
    }
    float m   = rs[0] * (1.f / CHW);
    float var = rq[0] * (1.f / CHW) - m * m;
    float inv = rsqrtf(var + 1e-5f);
    float* yp = y + (size_t)n * CHW;
    for (int i = threadIdx.x; i < CHW; i += 256) {
        int c = i >> 10;
        float v = (xp[i] - m) * inv * g[c] + bt[c];
        if (silu) v = v * sigm(v);
        yp[i] = v;
    }
}

// ------------- channel avg/max pool: [n,128,32,32] -> [n,2,32,32] -----------
__global__ __launch_bounds__(256)
void pool_kernel(const float* __restrict__ x, float* __restrict__ pooled)
{
    int n = blockIdx.y;
    int p = blockIdx.x * 256 + threadIdx.x;
    const float* xp = x + (size_t)n * CHW + p;
    float s = 0.f, mx = -1e30f;
    for (int c = 0; c < 128; ++c) { float v = xp[(size_t)c * HW]; s += v; mx = fmaxf(mx, v); }
    pooled[(size_t)n * 2048 + p]      = s * (1.f / 128.f);
    pooled[(size_t)n * 2048 + HW + p] = mx;
}

// ---------- 3x3 conv (2->1) + bias + sigmoid on pooled maps -----------------
__global__ __launch_bounds__(256)
void sa_kernel(const float* __restrict__ pooled, const float* __restrict__ w,
               const float* __restrict__ bias, float* __restrict__ sa)
{
    int n = blockIdx.y;
    int p = blockIdx.x * 256 + threadIdx.x;
    int y = p >> 5, x = p & 31;
    float acc = bias[0];
    for (int ch = 0; ch < 2; ++ch)
        for (int dy = 0; dy < 3; ++dy) {
            int yy = y + dy - 1; if (yy < 0 || yy > 31) continue;
            for (int dx = 0; dx < 3; ++dx) {
                int xx = x + dx - 1; if (xx < 0 || xx > 31) continue;
                acc += w[ch * 9 + dy * 3 + dx] *
                       pooled[(size_t)n * 2048 + ch * HW + yy * 32 + xx];
            }
        }
    sa[(size_t)n * HW + p] = sigm(acc);
}

// ------------ q_fg[n,c] = mean_p( k_sa[n,p] * qf[n,c,p] ) -------------------
__global__ __launch_bounds__(256)
void qfg_kernel(const float* __restrict__ qf, const float* __restrict__ ksa,
                float* __restrict__ qfg)
{
    __shared__ float sk[HW];
    int n = blockIdx.x;
    for (int i = threadIdx.x; i < HW; i += 256) sk[i] = ksa[(size_t)n * HW + i];
    __syncthreads();
    int wv = threadIdx.x >> 5, lane = threadIdx.x & 31;
    for (int cb = 0; cb < 16; ++cb) {
        int c = cb * 8 + wv;
        const float* qp = qf + (size_t)n * CHW + (size_t)c * HW;
        float s = 0.f;
        for (int p = lane; p < HW; p += 32) s += qp[p] * sk[p];
        for (int off = 16; off > 0; off >>= 1) s += __shfl_down(s, off);
        if (lane == 0) qfg[n * 128 + c] = s * (1.f / HW);
    }
}

// ----- s_prob[n,p] = sigmoid( sum_c q_fg[n,c] * kf[n,c,p] ) -----------------
__global__ __launch_bounds__(256)
void sprob_kernel(const float* __restrict__ kf, const float* __restrict__ qfg,
                  float* __restrict__ sprob)
{
    __shared__ float sg[128];
    int n = blockIdx.x;
    if (threadIdx.x < 128) sg[threadIdx.x] = qfg[n * 128 + threadIdx.x];
    __syncthreads();
    for (int p = threadIdx.x; p < HW; p += 256) {
        float acc = 0.f;
        const float* kp = kf + (size_t)n * CHW + p;
        for (int c = 0; c < 128; ++c) acc += kp[(size_t)c * HW] * sg[c];
        sprob[(size_t)n * HW + p] = sigm(acc);
    }
}

// - logits[l,s,b] = D^-1/2 * sum_d qf*q_sa * kf*k_sa  (D = 131072) -----------
__global__ __launch_bounds__(256)
void logits_kernel(const float* __restrict__ qf, const float* __restrict__ kf,
                   const float* __restrict__ qsa, const float* __restrict__ ksa,
                   float* __restrict__ logits)
{
    __shared__ float sq[HW], sk[HW], red[256];
    int b = blockIdx.x, s = blockIdx.y, l = blockIdx.z;
    int nq = l * 16 + b, nk = s * 16 + b;
    for (int i = threadIdx.x; i < HW; i += 256) {
        sq[i] = qsa[(size_t)nq * HW + i];
        sk[i] = ksa[(size_t)nk * HW + i];
    }
    __syncthreads();
    const float* qp = qf + (size_t)nq * CHW;
    const float* kp = kf + (size_t)nk * CHW;
    float acc = 0.f;
    for (int d = threadIdx.x; d < CHW; d += 256) {
        int p = d & 1023;
        acc += qp[d] * sq[p] * kp[d] * sk[p];
    }
    red[threadIdx.x] = acc; __syncthreads();
    for (int st = 128; st > 0; st >>= 1) {
        if (threadIdx.x < st) red[threadIdx.x] += red[threadIdx.x + st];
        __syncthreads();
    }
    if (threadIdx.x == 0)
        logits[(l * 8 + s) * 16 + b] = red[0] * 0.002762135864f; // 131072^-0.5
}

// ------- softmax over s per (l,b); fold sum over l: w_sum[s,b] --------------
__global__ void softmax_wsum_kernel(const float* __restrict__ logits,
                                    float* __restrict__ w_sum)
{
    __shared__ float accum[128];
    int tid = threadIdx.x;              // 128 threads: tid = l*16+b
    accum[tid] = 0.f;
    __syncthreads();
    int l = tid >> 4, b = tid & 15;
    float v[8], m = -1e30f;
    for (int s = 0; s < 8; ++s) { v[s] = logits[(l * 8 + s) * 16 + b]; m = fmaxf(m, v[s]); }
    float sum = 0.f;
    for (int s = 0; s < 8; ++s) { v[s] = expf(v[s] - m); sum += v[s]; }
    float inv = 1.f / sum;
    for (int s = 0; s < 8; ++s) atomicAdd(&accum[s * 16 + b], v[s] * inv);
    __syncthreads();
    w_sum[tid] = accum[tid];            // tid = s*16+b after reinterpretation
}

// --- out_pre[b,c,p] = sum_s w_sum[s,b] * vf[s,b,c,p] * s_prob[s,b,p] --------
__global__ __launch_bounds__(256)
void outpre_kernel(const float* __restrict__ vf, const float* __restrict__ sprob,
                   const float* __restrict__ wsum, float* __restrict__ outp)
{
    size_t idx = (size_t)blockIdx.x * 256 + threadIdx.x;   // 2,097,152 total
    int b   = (int)(idx >> 17);
    int rem = (int)(idx & (CHW - 1));
    int p   = (int)(idx & 1023);
    float acc = 0.f;
    for (int s = 0; s < 8; ++s) {
        int n = s * 16 + b;
        acc += wsum[n] * vf[(size_t)n * CHW + rem] * sprob[(size_t)n * HW + p];
    }
    outp[idx] = acc;
}

// ----- renormalize pos_table rows to max_norm = 1 ---------------------------
__global__ void pos_renorm_kernel(const float* __restrict__ pt, float* __restrict__ pr)
{
    __shared__ float sc[8];
    if (threadIdx.x < 8) {
        float s = 0.f;
        for (int j = 0; j < 256; ++j) { float v = pt[threadIdx.x * 256 + j]; s += v * v; }
        sc[threadIdx.x] = fminf(1.f, 1.f / fmaxf(sqrtf(s), 1e-7f));
    }
    __syncthreads();
    for (int i = threadIdx.x; i < 2048; i += 256) pr[i] = pt[i] * sc[i >> 8];
}

// ---------------------------------------------------------------------------
extern "C" void kernel_launch(void* const* d_in, const int* in_sizes, int n_in,
                              void* d_out, int out_size, void* d_ws, size_t ws_size,
                              hipStream_t stream)
{
    const float* q          = (const float*)d_in[0];
    const float* k          = (const float*)d_in[1];
    const float* v          = (const float*)d_in[2];
    const float* q_conv_w   = (const float*)d_in[3];
    const float* q_gn_w     = (const float*)d_in[4];
    const float* q_gn_b     = (const float*)d_in[5];
    const float* q_sa_w     = (const float*)d_in[6];
    const float* q_sa_b     = (const float*)d_in[7];
    const float* k_conv_w   = (const float*)d_in[8];
    const float* k_gn_w     = (const float*)d_in[9];
    const float* k_gn_b     = (const float*)d_in[10];
    const float* k_sa_w     = (const float*)d_in[11];
    const float* k_sa_b     = (const float*)d_in[12];
    const float* v_conv_w   = (const float*)d_in[13];
    const float* v_gn_w     = (const float*)d_in[14];
    const float* v_gn_b     = (const float*)d_in[15];
    const float* post_conv_w= (const float*)d_in[16];
    const float* post_gn_w  = (const float*)d_in[17];
    const float* post_gn_b  = (const float*)d_in[18];
    const float* pos_table  = (const float*)d_in[19];

    float* ws = (float*)d_ws;
    size_t o = 0;
    float* qf       = ws + o; o += (size_t)NS * CHW;      // 16.8M
    float* kf       = ws + o; o += (size_t)NS * CHW;      // 16.8M
    float* vf       = ws + o; o += (size_t)NS * CHW;      // 16.8M
    float* pooled_q = ws + o; o += (size_t)NS * 2 * HW;
    float* pooled_k = ws + o; o += (size_t)NS * 2 * HW;
    float* q_sa     = ws + o; o += (size_t)NS * HW;
    float* k_sa     = ws + o; o += (size_t)NS * HW;
    float* q_fg     = ws + o; o += (size_t)NS * 128;
    float* s_prob   = ws + o; o += (size_t)NS * HW;
    float* logitsB  = ws + o; o += 1024;
    float* w_sum    = ws + o; o += 128;
    float* pos_r    = ws + o; o += 2048;
    float* out_pre  = ws + o; o += (size_t)16 * CHW;
    float* out_conv = ws + o; o += (size_t)16 * CHW;

    constexpr size_t SH3 = (size_t)(9 * 128 * 32 + 6 * 34 * 32) * sizeof(bf16); // 86784 B
    constexpr size_t SH1 = 49152;                                               // TDM 1x1 conv

    pos_renorm_kernel<<<1, 256, 0, stream>>>(pos_table, pos_r);

    // q/k/v feature convs (WMMA implicit GEMM) + GroupNorm
    conv_wmma_kernel<128, 3><<<dim3(8, NS), 256, SH3, stream>>>(q, q_conv_w, qf, nullptr);
    gn_kernel<<<NS, 256, 0, stream>>>(qf, q_gn_w, q_gn_b, qf, 0);

    conv_wmma_kernel<256, 3><<<dim3(8, NS), 256, SH3, stream>>>(k, k_conv_w, kf, nullptr);
    gn_kernel<<<NS, 256, 0, stream>>>(kf, k_gn_w, k_gn_b, kf, 0);

    conv_wmma_kernel<256, 3><<<dim3(8, NS), 256, SH3, stream>>>(v, v_conv_w, vf, pos_r);
    gn_kernel<<<NS, 256, 0, stream>>>(vf, v_gn_w, v_gn_b, vf, 0);

    // spatial attention filters
    pool_kernel<<<dim3(4, NS), 256, 0, stream>>>(qf, pooled_q);
    sa_kernel<<<dim3(4, NS), 256, 0, stream>>>(pooled_q, q_sa_w, q_sa_b, q_sa);
    pool_kernel<<<dim3(4, NS), 256, 0, stream>>>(kf, pooled_k);
    sa_kernel<<<dim3(4, NS), 256, 0, stream>>>(pooled_k, k_sa_w, k_sa_b, k_sa);

    // spatial sigmoid gate
    qfg_kernel<<<NS, 256, 0, stream>>>(qf, k_sa, q_fg);
    sprob_kernel<<<NS, 256, 0, stream>>>(kf, q_fg, s_prob);

    // temporal attention (softmax over s; sum over l folded into w_sum)
    logits_kernel<<<dim3(16, 8, 8), 256, 0, stream>>>(qf, kf, q_sa, k_sa, logitsB);
    softmax_wsum_kernel<<<1, 128, 0, stream>>>(logitsB, w_sum);
    outpre_kernel<<<8192, 256, 0, stream>>>(vf, s_prob, w_sum, out_pre);

    // post 1x1 conv (WMMA + TDM async tensor load) + GN + SiLU -> d_out
    conv1x1_wmma_tdm_kernel<<<dim3(8, 16), 256, SH1, stream>>>(out_pre, post_conv_w, out_conv);
    gn_kernel<<<16, 256, 0, stream>>>(out_conv, post_gn_w, post_gn_b, (float*)d_out, 1);
}
